// MultiHeadLatentAttention_28578712387976
// MI455X (gfx1250) — matrix operand order repaired and checked by reading, not compiled, not run on hardware
//
#include <hip/hip_runtime.h>
#include <hip/hip_bf16.h>
#include <cstdint>
#include <math.h>

// ---------------------------------------------------------------------------
// MLA forward for MI455X (gfx1250, wave32).
//  - All GEMM FLOPs through v_wmma_f32_16x16x32_bf16 (fp32 accumulate).
//  - Weights pre-transposed at cast time so every WMMA operand (A and B) is
//    K-contiguous per lane -> ds_load_b128 pairs, no v_mov_b16 repacking.
//  - V written transposed by the value GEMM (vector bf16x8 stores) so the
//    fused masked-softmax+P*V kernel also gets contiguous B fragments.
//  - Scores / attn_weights stay fp32 (outputs; bandwidth-bound).
// Workspace requirement: ~130 MB.
// ---------------------------------------------------------------------------

typedef __bf16 bf16_t;
typedef __attribute__((ext_vector_type(16))) __bf16 v16bf;
typedef __attribute__((ext_vector_type(8)))  float  v8f;

constexpr int CB  = 2;      // batch
constexpr int CL  = 1024;   // query length
constexpr int CLP = 1024;   // prefix kv length
constexpr int CD  = 2048;   // model dim
constexpr int CDL = 1024;   // latent dim
constexpr int CH  = 16;     // heads
constexpr int CDH = 64;     // v head dim
constexpr int CKV = CLP + CL;   // 2048
constexpr int CHD = CH * CDH;   // 1024
constexpr int CQK = 2 * CHD;    // 2048 (q_full/k_full width)

// ---------------- WMMA helpers (ISA 7.12.2 bf16 layouts) -------------------

__device__ inline v8f wmma_bf16(v16bf a, v16bf b, v8f c) {
  return __builtin_amdgcn_wmma_f32_16x16x32_bf16(
      /*neg_a=*/false, a, /*neg_b=*/false, b,
      /*c_mod=*/(short)0, c, /*reuse_a=*/false, /*reuse_b=*/false);
}

// A fragment: 16(M) x 32(K). lanes 0-15: M=lane, K={0..7,16..23};
// lanes 16-31: M=lane-16, K={8..15,24..31}. Source row-major (M x K).
// Per lane: two contiguous 8x bf16 runs -> 2x ds_load_b128.
__device__ inline v16bf load_a_frag(const bf16_t* p, int ld) {
  const int lane = threadIdx.x & 31;
  const bf16_t* r = p + (lane & 15) * ld + ((lane >> 4) << 3);
  v16bf f;
#pragma unroll
  for (int e = 0; e < 8; ++e) f[e] = r[e];
#pragma unroll
  for (int e = 0; e < 8; ++e) f[8 + e] = r[16 + e];
  return f;
}

// B fragment from B^T storage (N x K row-major): lanes 0-15 K={0..7,16..23},
// lanes 16-31 K={8..15,24..31} (matches A) -> 2x ds_load_b128.
__device__ inline v16bf load_bt_frag(const bf16_t* p, int ld) {
  const int lane = threadIdx.x & 31;
  const bf16_t* r = p + (lane & 15) * ld + ((lane >> 4) << 3);
  v16bf f;
#pragma unroll
  for (int e = 0; e < 8; ++e) f[e] = r[e];
#pragma unroll
  for (int e = 0; e < 8; ++e) f[8 + e] = r[16 + e];
  return f;
}

// ---------------- Generic batched GEMM vs pre-transposed B -----------------
// C[m,n] = scale * sum_k A[m,k]*Bt[n,k] + bias[n]
// 256 threads = 8 waves; block tile 256(M) x 128(N); K-step 32;
// wave tile 64x64 -> 4x4 WMMA accumulators; 16 WMMA per 16 ds_load_b128.
__global__ __launch_bounds__(256) void gemm_nt_kernel(
    const bf16_t* __restrict__ A, int lda, long long strideA,
    const bf16_t* __restrict__ Bt, int ldbt,
    float* __restrict__ Cf, bf16_t* __restrict__ Cb, int ldc, long long strideC,
    const float* __restrict__ bias, int K, float scale, int transC) {
  __shared__ __align__(16) bf16_t As[256][48];   // 24 KB (row stride 96B)
  __shared__ __align__(16) bf16_t Bs[128][48];   // 12 KB
  const int tid  = threadIdx.x;
  const int lane = tid & 31;
  const int wave = tid >> 5;
  const int wm = (wave >> 1) * 64;   // 4 waves along M
  const int wn = (wave & 1) * 64;    // 2 waves along N
  const int m0 = blockIdx.y * 256;
  const int n0 = blockIdx.x * 128;
  const bf16_t* Ab = A + (long long)blockIdx.z * strideA;
  const long long cbase = (long long)blockIdx.z * strideC;

  const int rowB = tid >> 1, halfB = (tid & 1) * 16;   // 128x32 Bt tile

  v8f acc[4][4];
#pragma unroll
  for (int i = 0; i < 4; ++i)
#pragma unroll
    for (int j = 0; j < 4; ++j)
#pragma unroll
      for (int r = 0; r < 8; ++r) acc[i][j][r] = 0.0f;

  for (int k0 = 0; k0 < K; k0 += 32) {
    // A: 256 rows x 32 -> each thread owns one full row segment (4x uint4).
    const uint4* ga =
        reinterpret_cast<const uint4*>(Ab + (long long)(m0 + tid) * lda + k0);
    uint4 a0 = ga[0], a1 = ga[1], a2 = ga[2], a3 = ga[3];
    // Bt: 128 rows x 32 -> two threads per row (2x uint4 each).
    const uint4* gb = reinterpret_cast<const uint4*>(
        Bt + (long long)(n0 + rowB) * ldbt + k0 + halfB);
    uint4 b0 = gb[0], b1 = gb[1];
    if (k0 + 32 < K) {  // gfx1250 global_prefetch_b8
      __builtin_prefetch(Ab + (long long)(m0 + tid) * lda + (k0 + 32), 0, 1);
      __builtin_prefetch(Bt + (long long)(n0 + rowB) * ldbt + (k0 + 32) + halfB, 0, 1);
    }
    __syncthreads();
    {
      uint4* da = reinterpret_cast<uint4*>(&As[tid][0]);
      da[0] = a0; da[1] = a1;
      uint4* da2 = reinterpret_cast<uint4*>(&As[tid][16]);
      da2[0] = a2; da2[1] = a3;
      uint4* db = reinterpret_cast<uint4*>(&Bs[rowB][halfB]);
      db[0] = b0; db[1] = b1;
    }
    __syncthreads();
    v16bf af[4];
#pragma unroll
    for (int i = 0; i < 4; ++i) af[i] = load_a_frag(&As[wm + i * 16][0], 48);
#pragma unroll
    for (int j = 0; j < 4; ++j) {
      const v16bf bv = load_bt_frag(&Bs[wn + j * 16][0], 48);
#pragma unroll
      for (int i = 0; i < 4; ++i) acc[i][j] = wmma_bf16(af[i], bv, acc[i][j]);
    }
  }

  // Epilogue. C layout: VGPR r -> M = rofs + r; N = col.
  const int col = lane & 15, rofs = (lane >> 4) * 8;
#pragma unroll
  for (int i = 0; i < 4; ++i) {
    const int gm0 = m0 + wm + i * 16 + rofs;
#pragma unroll
    for (int j = 0; j < 4; ++j) {
      const int gn = n0 + wn + j * 16 + col;
      const float bvv = bias ? bias[gn] : 0.0f;
      if (transC && Cb) {
        // C^T store: 8 consecutive M values -> one packed bf16x8 store.
        alignas(16) bf16_t pk[8];
#pragma unroll
        for (int r = 0; r < 8; ++r)
          pk[r] = (bf16_t)(acc[i][j][r] * scale + bvv);
        *reinterpret_cast<uint4*>(Cb + cbase + (long long)gn * ldc + gm0) =
            *reinterpret_cast<const uint4*>(pk);
      } else {
#pragma unroll
        for (int r = 0; r < 8; ++r) {
          const float v = acc[i][j][r] * scale + bvv;
          const long long off = cbase + (long long)(gm0 + r) * ldc + gn;
          if (Cf) Cf[off] = v;
          if (Cb) Cb[off] = (bf16_t)v;
        }
      }
    }
  }
}

// ---------------- Scores: S[b,h,l,kv] = (q.k)/sqrt(128) --------------------
// k_full rows are (kv x 128-per-head) -> already N x K contiguous for B^T.
__global__ __launch_bounds__(256) void scores_kernel(
    const bf16_t* __restrict__ Qf, const bf16_t* __restrict__ Kf,
    float* __restrict__ Sc, float scale) {
  __shared__ __align__(16) bf16_t As[256][48];
  __shared__ __align__(16) bf16_t Bs[128][48];
  const int tid = threadIdx.x, lane = tid & 31, wave = tid >> 5;
  const int wm = (wave >> 1) * 64, wn = (wave & 1) * 64;
  const int z = blockIdx.z, bb = z / CH, h = z % CH;
  const int m0 = blockIdx.y * 256;   // query rows
  const int n0 = blockIdx.x * 128;   // key rows
  const bf16_t* Qb = Qf + ((long long)bb * CL) * CQK + h * 128;
  const bf16_t* Kb = Kf + ((long long)bb * CKV) * CQK + h * 128;
  float* Cp = Sc + ((long long)z * CL) * CKV;
  const int rowB = tid >> 1, halfB = (tid & 1) * 16;

  v8f acc[4][4];
#pragma unroll
  for (int i = 0; i < 4; ++i)
#pragma unroll
    for (int j = 0; j < 4; ++j)
#pragma unroll
      for (int r = 0; r < 8; ++r) acc[i][j][r] = 0.0f;

  for (int k0 = 0; k0 < 128; k0 += 32) {
    const uint4* ga =
        reinterpret_cast<const uint4*>(Qb + (long long)(m0 + tid) * CQK + k0);
    uint4 a0 = ga[0], a1 = ga[1], a2 = ga[2], a3 = ga[3];
    const uint4* gk = reinterpret_cast<const uint4*>(
        Kb + (long long)(n0 + rowB) * CQK + k0 + halfB);
    uint4 b0 = gk[0], b1 = gk[1];
    __syncthreads();
    {
      uint4* da = reinterpret_cast<uint4*>(&As[tid][0]);
      da[0] = a0; da[1] = a1;
      uint4* da2 = reinterpret_cast<uint4*>(&As[tid][16]);
      da2[0] = a2; da2[1] = a3;
      uint4* db = reinterpret_cast<uint4*>(&Bs[rowB][halfB]);
      db[0] = b0; db[1] = b1;
    }
    __syncthreads();
    v16bf af[4];
#pragma unroll
    for (int i = 0; i < 4; ++i) af[i] = load_a_frag(&As[wm + i * 16][0], 48);
#pragma unroll
    for (int j = 0; j < 4; ++j) {
      const v16bf bv = load_bt_frag(&Bs[wn + j * 16][0], 48);
#pragma unroll
      for (int i = 0; i < 4; ++i) acc[i][j] = wmma_bf16(af[i], bv, acc[i][j]);
    }
  }

  const int col = lane & 15, rofs = (lane >> 4) * 8;
#pragma unroll
  for (int i = 0; i < 4; ++i)
#pragma unroll
    for (int j = 0; j < 4; ++j)
#pragma unroll
      for (int r = 0; r < 8; ++r) {
        const int gm = m0 + wm + i * 16 + rofs + r;
        const int gn = n0 + wn + j * 16 + col;
        Cp[(long long)gm * CKV + gn] = acc[i][j][r] * scale;
      }
}

// ---------------- Unmasked row softmax (attn_weights) ----------------------
__global__ __launch_bounds__(256) void softmax_kernel(
    const float* __restrict__ S, float* __restrict__ O) {
  const long long row = blockIdx.x;
  const float* s = S + row * (long long)CKV;
  float* o = O + row * (long long)CKV;
  __shared__ float red[256];
  const int tid = threadIdx.x;
  float mx = -1e30f;
  for (int c = tid; c < CKV; c += 256) mx = fmaxf(mx, s[c]);
  red[tid] = mx; __syncthreads();
  for (int w = 128; w > 0; w >>= 1) {
    if (tid < w) red[tid] = fmaxf(red[tid], red[tid + w]);
    __syncthreads();
  }
  mx = red[0]; __syncthreads();
  float sm = 0.0f;
  for (int c = tid; c < CKV; c += 256) sm += __expf(s[c] - mx);
  red[tid] = sm; __syncthreads();
  for (int w = 128; w > 0; w >>= 1) {
    if (tid < w) red[tid] += red[tid + w];
    __syncthreads();
  }
  const float inv = 1.0f / red[0];
  for (int c = tid; c < CKV; c += 256) o[c] = __expf(s[c] - mx) * inv;
}

// ---------------- Fused masked softmax + P*V (WMMA) ------------------------
// Block: 128 query rows of one (b,h); wave w owns 16 rows. Probs generated on
// the fly into bf16 A-fragments; V^T tile staged in LDS (contiguous B frags).
__global__ __launch_bounds__(256) void pv_kernel(
    const float* __restrict__ S, const bf16_t* __restrict__ VT,
    const int* __restrict__ vlens, bf16_t* __restrict__ Hb) {
  __shared__ float smax[128], sinv[128];
  __shared__ int svl[128];
  __shared__ __align__(16) bf16_t Vts[64][48];   // dh x kv tile (transposed V)
  const int tid = threadIdx.x, lane = tid & 31, wave = tid >> 5;
  const int z = blockIdx.z, bb = z / CH, h = z % CH;
  const int m0 = blockIdx.x * 128;
  const float* Sb = S + ((long long)z * CL) * CKV;
  // VT: (B, CHD, CKV); rows = head-dim channels, contiguous along kv.
  const bf16_t* Vp = VT + (long long)bb * CHD * CKV + (long long)(h * CDH) * CKV;

  for (int r = tid; r < 128; r += 256) svl[r] = vlens[bb * CL + m0 + r];
  __syncthreads();

  // Phase 1: masked row max / sum (one wave per 16 rows).
  for (int rr = 0; rr < 16; ++rr) {
    const int row = wave * 16 + rr;
    const int vl = svl[row];
    const float* sr = Sb + (long long)(m0 + row) * CKV;
    float mx = -1e30f;
    for (int c = lane; c < CKV; c += 32) {
      const float v = (c >= vl) ? -10000.0f : sr[c];
      mx = fmaxf(mx, v);
    }
#pragma unroll
    for (int o = 16; o > 0; o >>= 1) mx = fmaxf(mx, __shfl_xor(mx, o, 32));
    float sm = 0.0f;
    for (int c = lane; c < CKV; c += 32) {
      const float v = (c >= vl) ? -10000.0f : sr[c];
      sm += __expf(v - mx);
    }
#pragma unroll
    for (int o = 16; o > 0; o >>= 1) sm += __shfl_xor(sm, o, 32);
    if (lane == 0) { smax[row] = mx; sinv[row] = 1.0f / sm; }
  }
  __syncthreads();

  // Phase 2: stream KV in 32-col tiles.
  const int arow = lane & 15, khalf = (lane >> 4) * 8;
  const int myrow = wave * 16 + arow;
  const float mx = smax[myrow], inv = sinv[myrow];
  const int vl = svl[myrow];
  const float* sr = Sb + (long long)(m0 + myrow) * CKV;
  v8f acc[4];
#pragma unroll
  for (int j = 0; j < 4; ++j)
#pragma unroll
    for (int r = 0; r < 8; ++r) acc[j][r] = 0.0f;

  const int vr = tid >> 2, vc = (tid & 3) * 8;   // 64x32 V^T tile, uint4/thread
  for (int kv0 = 0; kv0 < CKV; kv0 += 32) {
    const uint4 vt =
        *reinterpret_cast<const uint4*>(Vp + (long long)vr * CKV + kv0 + vc);
    __syncthreads();
    *reinterpret_cast<uint4*>(&Vts[vr][vc]) = vt;
    __syncthreads();
    v16bf af;
#pragma unroll
    for (int e = 0; e < 16; ++e) {
      const int k = kv0 + ((e < 8) ? (khalf + e) : (16 + khalf + (e - 8)));
      const float v = (k >= vl) ? -10000.0f : sr[k];
      af[e] = (bf16_t)(__expf(v - mx) * inv);
    }
#pragma unroll
    for (int j = 0; j < 4; ++j) {
      const v16bf bv = load_bt_frag(&Vts[j * 16][0], 48);
      acc[j] = wmma_bf16(af, bv, acc[j]);
    }
  }

  const int col = lane & 15, rofs = (lane >> 4) * 8;
#pragma unroll
  for (int j = 0; j < 4; ++j)
#pragma unroll
    for (int r = 0; r < 8; ++r) {
      const int gm = m0 + wave * 16 + rofs + r;
      Hb[((long long)bb * CL + gm) * CHD + h * CDH + j * 16 + col] =
          (bf16_t)acc[j][r];
    }
}

// ---------------- Elementwise / data-marshalling kernels -------------------

__global__ void cast_bf16_kernel(const float* __restrict__ in,
                                 bf16_t* __restrict__ out, long long n) {
  long long i = (long long)blockIdx.x * blockDim.x + threadIdx.x;
  const long long st = (long long)gridDim.x * blockDim.x;
  for (; i < n; i += st) out[i] = (bf16_t)in[i];
}

// f32 (K x N) -> bf16 (N x K), LDS-tiled 32x32.
__global__ __launch_bounds__(256) void cast_transpose_kernel(
    const float* __restrict__ in, bf16_t* __restrict__ out, int K, int N) {
  __shared__ float tile[32][33];
  const int k0 = blockIdx.y * 32, n0 = blockIdx.x * 32;
  const int tx = threadIdx.x & 31, ty = threadIdx.x >> 5;
  for (int r = ty; r < 32; r += 8)
    tile[r][tx] = in[(long long)(k0 + r) * N + n0 + tx];
  __syncthreads();
  for (int r = ty; r < 32; r += 8)
    out[(long long)(n0 + r) * K + k0 + tx] = (bf16_t)tile[tx][r];
}

// Copy (CB, per_batch) contiguous input into batched output at b*out_stride,
// writing both f32 (d_out region) and bf16 (workspace).
__global__ void copy_prefix_kernel(const float* __restrict__ in,
                                   float* __restrict__ outf,
                                   bf16_t* __restrict__ outb,
                                   long long per_batch, long long out_stride) {
  const long long n = (long long)CB * per_batch;
  long long i = (long long)blockIdx.x * blockDim.x + threadIdx.x;
  const long long st = (long long)gridDim.x * blockDim.x;
  for (; i < n; i += st) {
    const long long b = i / per_batch, r = i % per_batch;
    const float v = in[i];
    const long long o = b * out_stride + r;
    outf[o] = v;
    outb[o] = (bf16_t)v;
  }
}

// q_full = [query | rotary(q_rope_raw)] (plain concat; head split is just the
// h*128 column offset used by the score kernel).
__global__ void pack_qfull_kernel(const bf16_t* __restrict__ queryb,
                                  const float* __restrict__ qrope,
                                  bf16_t* __restrict__ qfull) {
  const long long n = (long long)CB * CL * CHD;
  long long i = (long long)blockIdx.x * blockDim.x + threadIdx.x;
  const long long st = (long long)gridDim.x * blockDim.x;
  for (; i < n; i += st) {
    const int c = (int)(i % CHD);
    const long long bl = i / CHD;
    const int l = (int)(bl % CL);
    qfull[bl * CQK + c] = queryb[i];
    const int h = c / CDH, d = c % CDH, j = d & 31;
    const float invf = __powf(10000.0f, -(float)(2 * j) / (float)CDH);
    const float ang = (float)l * invf;
    const float cs = __cosf(ang), sn = __sinf(ang);
    const float x1 = qrope[bl * CHD + h * CDH + j];
    const float x2 = qrope[bl * CHD + h * CDH + 32 + j];
    const float v = (d < 32) ? (x1 * cs - x2 * sn) : (x1 * sn + x2 * cs);
    qfull[bl * CQK + CHD + c] = (bf16_t)v;
  }
}

// rotary(k_rope_raw) -> key_rope_all rows [LP, LP+L): f32 output + bf16 copy.
__global__ void rotary_k_kernel(const float* __restrict__ krope,
                                float* __restrict__ kro_f,
                                bf16_t* __restrict__ kro_b) {
  const long long n = (long long)CB * CL * CHD;
  long long i = (long long)blockIdx.x * blockDim.x + threadIdx.x;
  const long long st = (long long)gridDim.x * blockDim.x;
  for (; i < n; i += st) {
    const int c = (int)(i % CHD);
    const long long bl = i / CHD;
    const int b = (int)(bl / CL), l = (int)(bl % CL);
    const int h = c / CDH, d = c % CDH, j = d & 31;
    const float invf = __powf(10000.0f, -(float)(2 * j) / (float)CDH);
    const float ang = (float)l * invf;
    const float cs = __cosf(ang), sn = __sinf(ang);
    const float x1 = krope[bl * CHD + h * CDH + j];
    const float x2 = krope[bl * CHD + h * CDH + 32 + j];
    const float v = (d < 32) ? (x1 * cs - x2 * sn) : (x1 * sn + x2 * cs);
    const long long o = ((long long)b * CKV + CLP + l) * CHD + c;
    kro_f[o] = v;
    kro_b[o] = (bf16_t)v;
  }
}

// k_full = [key_c | key_rope_all] (bf16 concat).
__global__ void pack_kfull_kernel(const bf16_t* __restrict__ keycb,
                                  const bf16_t* __restrict__ krob,
                                  bf16_t* __restrict__ kfull) {
  const long long n = (long long)CB * CKV * CHD;
  long long i = (long long)blockIdx.x * blockDim.x + threadIdx.x;
  const long long st = (long long)gridDim.x * blockDim.x;
  for (; i < n; i += st) {
    const int c = (int)(i % CHD);
    const long long row = i / CHD;
    kfull[row * CQK + c] = keycb[i];
    kfull[row * CQK + CHD + c] = krob[i];
  }
}

// ---------------------------------------------------------------------------

extern "C" void kernel_launch(void* const* d_in, const int* in_sizes, int n_in,
                              void* d_out, int out_size, void* d_ws,
                              size_t ws_size, hipStream_t stream) {
  const float* x         = (const float*)d_in[0];
  const float* z         = (const float*)d_in[1];
  const float* key_rope  = (const float*)d_in[2];
  const int*   valid_len = (const int*)d_in[3];
  const float* W_latent  = (const float*)d_in[4];
  const float* W_q_down  = (const float*)d_in[5];
  const float* b_q_down  = (const float*)d_in[6];
  const float* W_q_up    = (const float*)d_in[7];
  const float* W_k_up    = (const float*)d_in[8];
  const float* W_v_up    = (const float*)d_in[9];
  const float* W_x_rope  = (const float*)d_in[10];
  const float* W_k_rope  = (const float*)d_in[11];
  const float* W_o       = (const float*)d_in[12];

  // d_out layout (flat concat, return order)
  float* out    = (float*)d_out;                         // (B,L,D)
  float* zall_f = out + (long long)CB * CL * CD;         // (B,KV,DL)
  float* kro_f  = zall_f + (long long)CB * CKV * CDL;    // (B,KV,H*DH)
  float* attn_w = kro_f + (long long)CB * CKV * CHD;     // (B,H,L,KV)
  float* scores = attn_w + (long long)CB * CH * CL * CKV;

  // workspace carve-up (256B aligned)
  char* wp = (char*)d_ws;
  auto alloc = [&](size_t bytes) -> void* {
    void* r = (void*)wp;
    wp += (bytes + 255) & ~(size_t)255;
    return r;
  };
  bf16_t* xb     = (bf16_t*)alloc((size_t)CB * CL * CD * 2);
  bf16_t* wlatT  = (bf16_t*)alloc((size_t)CD * CDL * 2);   // (DL x D)
  bf16_t* wqdT   = (bf16_t*)alloc((size_t)CD * CDL * 2);   // (DL x D)
  bf16_t* wquT   = (bf16_t*)alloc((size_t)CDL * CHD * 2);  // (HD x DL)
  bf16_t* wkuT   = (bf16_t*)alloc((size_t)CDL * CHD * 2);
  bf16_t* wvuT   = (bf16_t*)alloc((size_t)CDL * CHD * 2);
  bf16_t* wxrT   = (bf16_t*)alloc((size_t)CDL * CHD * 2);
  bf16_t* wkrT   = (bf16_t*)alloc((size_t)CD * CHD * 2);   // (HD x D)
  bf16_t* woT    = (bf16_t*)alloc((size_t)CHD * CD * 2);   // (D x HD)
  bf16_t* zallb  = (bf16_t*)alloc((size_t)CB * CKV * CDL * 2);
  bf16_t* qlatb  = (bf16_t*)alloc((size_t)CB * CL * CDL * 2);
  bf16_t* queryb = (bf16_t*)alloc((size_t)CB * CL * CHD * 2);
  bf16_t* keycb  = (bf16_t*)alloc((size_t)CB * CKV * CHD * 2);
  bf16_t* vT     = (bf16_t*)alloc((size_t)CB * CHD * CKV * 2);  // V^T
  bf16_t* krob   = (bf16_t*)alloc((size_t)CB * CKV * CHD * 2);
  bf16_t* qfullb = (bf16_t*)alloc((size_t)CB * CL * CQK * 2);
  bf16_t* kfullb = (bf16_t*)alloc((size_t)CB * CKV * CQK * 2);
  bf16_t* headb  = (bf16_t*)alloc((size_t)CB * CL * CHD * 2);
  float*  qrr    = (float*)alloc((size_t)CB * CL * CHD * 4);
  float*  krr    = (float*)alloc((size_t)CB * CL * CHD * 4);

  cast_bf16_kernel<<<16384, 256, 0, stream>>>(x, xb, (long long)CB * CL * CD);

  auto castT = [&](const float* in, bf16_t* ot, int K, int N) {
    cast_transpose_kernel<<<dim3(N / 32, K / 32), 256, 0, stream>>>(in, ot, K, N);
  };
  castT(W_latent, wlatT, CD, CDL);
  castT(W_q_down, wqdT, CD, CDL);
  castT(W_q_up,  wquT, CDL, CHD);
  castT(W_k_up,  wkuT, CDL, CHD);
  castT(W_v_up,  wvuT, CDL, CHD);
  castT(W_x_rope, wxrT, CDL, CHD);
  castT(W_k_rope, wkrT, CD, CHD);
  castT(W_o, woT, CHD, CD);

  // z -> z_all[:, :LP, :]; key_rope -> key_rope_all[:, :LP, :]
  copy_prefix_kernel<<<8192, 256, 0, stream>>>(z, zall_f, zallb,
                                               (long long)CLP * CDL,
                                               (long long)CKV * CDL);
  copy_prefix_kernel<<<8192, 256, 0, stream>>>(key_rope, kro_f, krob,
                                               (long long)CLP * CHD,
                                               (long long)CKV * CHD);

  auto gemm = [&](const bf16_t* A, int lda, long long sA, const bf16_t* Bt,
                  int ldbt, float* Cf, bf16_t* Cbf, int ldc, long long sC,
                  const float* bias, int M, int N, int K, float scale,
                  int transC) {
    gemm_nt_kernel<<<dim3(N / 128, M / 256, CB), 256, 0, stream>>>(
        A, lda, sA, Bt, ldbt, Cf, Cbf, ldc, sC, bias, K, scale, transC);
  };

  // z_all[:, LP:, :] = x @ W_latent   (f32 output + bf16 copy)
  gemm(xb, CD, (long long)CL * CD, wlatT, CD,
       zall_f + (long long)CLP * CDL, zallb + (long long)CLP * CDL,
       CDL, (long long)CKV * CDL, nullptr, CL, CDL, CD, 1.0f, 0);
  // q_lat = x @ W_q_down + b_q_down   (bf16)
  gemm(xb, CD, (long long)CL * CD, wqdT, CD, nullptr, qlatb, CDL,
       (long long)CL * CDL, b_q_down, CL, CDL, CD, 1.0f, 0);
  // query = q_lat @ W_q_up            (bf16)
  gemm(qlatb, CDL, (long long)CL * CDL, wquT, CDL, nullptr, queryb, CHD,
       (long long)CL * CHD, nullptr, CL, CHD, CDL, 1.0f, 0);
  // q_rope_raw = q_lat @ W_x_rope     (f32, rotary input)
  gemm(qlatb, CDL, (long long)CL * CDL, wxrT, CDL, qrr, nullptr, CHD,
       (long long)CL * CHD, nullptr, CL, CHD, CDL, 1.0f, 0);
  // k_rope_raw = x @ W_k_rope         (f32)
  gemm(xb, CD, (long long)CL * CD, wkrT, CD, krr, nullptr, CHD,
       (long long)CL * CHD, nullptr, CL, CHD, CD, 1.0f, 0);
  // key_c = z_all @ W_k_up            (bf16), M = KV
  gemm(zallb, CDL, (long long)CKV * CDL, wkuT, CDL, nullptr, keycb, CHD,
       (long long)CKV * CHD, nullptr, CKV, CHD, CDL, 1.0f, 0);
  // value^T = (z_all @ W_v_up)^T      (bf16, packed uint4 stores)
  gemm(zallb, CDL, (long long)CKV * CDL, wvuT, CDL, nullptr, vT, CKV,
       (long long)CHD * CKV, nullptr, CKV, CHD, CDL, 1.0f, 1);
  // out = head_out @ W_o              (f32) -- launched after pv below

  pack_qfull_kernel<<<8192, 256, 0, stream>>>(queryb, qrr, qfullb);
  rotary_k_kernel<<<8192, 256, 0, stream>>>(krr, kro_f, krob);
  pack_kfull_kernel<<<16384, 256, 0, stream>>>(keycb, krob, kfullb);

  const float iscale = 0.0883883476483184f;  // 1/sqrt(2*DH)
  scores_kernel<<<dim3(CKV / 128, CL / 256, CB * CH), 256, 0, stream>>>(
      qfullb, kfullb, scores, iscale);
  softmax_kernel<<<dim3(CB * CH * CL), 256, 0, stream>>>(scores, attn_w);
  pv_kernel<<<dim3(CL / 128, 1, CB * CH), 256, 0, stream>>>(scores, vT,
                                                            valid_len, headb);
  // out = head_out @ W_o (f32)
  gemm(headb, CHD, (long long)CL * CHD, woT, CHD, out, nullptr, CD,
       (long long)CL * CD, nullptr, CL, CD, CHD, 1.0f, 0);
}
